// VLMSDeformAttnLayer_87746181857981
// MI455X (gfx1250) — matrix operand order repaired and checked by reading, not compiled, not run on hardware
//
#include <hip/hip_runtime.h>
#include <hip/hip_bf16.h>
#include <cstddef>

// ---------------------------------------------------------------------------
// VLMSDeformAttnLayer forward for MI455X (gfx1250, wave32, WMMA).
// All dense projections run through one f16-WMMA GEMM kernel
// (v_wmma_f32_16x16x32_f16), fp32 accumulate, fused bias/scale/relu/residual.
// Block tile 32(M) x 128(N), each wave owns a 16x32 D tile (2 accumulators,
// A-fragment reused across 2 B-fragments -> 4 static v_wmma in the K loop).
// Attention query rows are staged into LDS with global_load_async_to_lds_b128
// (ASYNCcnt path). Per-batch blocking keeps K/V slices (22 MB) in the 192 MB L2.
//
// Assumed input flattening order (setup_inputs dict order, params pytree in
// insertion order): 0 vis, 1 lang, 2 lang_mask, 3 vis_pos, 4 vis_padding_mask,
// 5 reference_points, 6 spatial_shapes, 7 level_start_index, then 52 param
// leaves starting at index 8. Workspace requirement ~250 MB.
// ---------------------------------------------------------------------------

#define BS   8
#define SQL  20            // language sequence
#define LTOT 5440          // vision tokens
#define E    256
#define AD   1024
#define HD   128
#define NH   8
#define VHD  32
#define CLAMPV 50000.0f
#define QSCALE 0.08838834764831845f   // 128^-0.5

typedef _Float16 half4 __attribute__((ext_vector_type(4)));
typedef _Float16 half8 __attribute__((ext_vector_type(8)));
typedef _Float16 v16h  __attribute__((ext_vector_type(16)));
typedef float    v8f   __attribute__((ext_vector_type(8)));

union AFrag { v16h v; half8 h[2]; };

// ---------------------------------------------------------------------------
// WMMA GEMM: C[M,N] = act( scale*(A[M,K] @ B[K,N] + bias) ) + resid
// Block tile 32(M) x 128(N), K-tile 64, 8 waves as 2(M) x 4(N),
// wave tile 16x32 => 2 v8f accumulators, 4 wmma per K-tile per wave.
// ---------------------------------------------------------------------------
__global__ __launch_bounds__(256)
void gemm_f16_wmma(const float* __restrict__ A, const float* __restrict__ B,
                   const float* __restrict__ bias, const float* __restrict__ resid,
                   float* __restrict__ C, int M, int N, int K,
                   float scale, int relu)
{
  __shared__ _Float16 As[32 * 72];    // 32 rows x 64 K (stride 72 halves = 144B)
  __shared__ _Float16 Bt[128 * 72];   // 128 N-rows x 64 K, B stored transposed

  const int tid  = threadIdx.x;
  const int wave = tid >> 5;
  const int lane = tid & 31;
  const int wm   = wave & 1;          // 2 wave-rows (M)
  const int wn   = wave >> 1;         // 4 wave-cols (N), 32 cols each
  const int m0   = blockIdx.x * 32;
  const int n0   = blockIdx.y * 128;

  v8f acc0 = {};
  v8f acc1 = {};

  for (int k0 = 0; k0 < K; k0 += 64) {
    // Prefetch next K-tile (global_prefetch_b8 on gfx1250).
    if (k0 + 64 < K) {
      __builtin_prefetch(A + (size_t)(m0 + (tid & 31)) * K + k0 + 64 + (tid >> 5) * 8, 0, 1);
      __builtin_prefetch(B + (size_t)(k0 + 64 + (tid & 63)) * N + n0 + (tid >> 6) * 32, 0, 1);
    }
    __syncthreads();

    // Stage A tile: 32x64 f32 -> f16, coalesced float4 reads along K.
#pragma unroll
    for (int j = 0; j < 2; ++j) {
      int e  = (tid + 256 * j) * 4;
      int kl = e & 63, ml = e >> 6;
      const float4 f = *(const float4*)(A + (size_t)(m0 + ml) * K + k0 + kl);
      half4 hv; hv[0] = (_Float16)f.x; hv[1] = (_Float16)f.y;
      hv[2] = (_Float16)f.z; hv[3] = (_Float16)f.w;
      *(half4*)&As[ml * 72 + kl] = hv;
    }
    // Stage B tile transposed: 64(K)x128(N) f32 -> Bt[n][k] f16.
#pragma unroll
    for (int j = 0; j < 8; ++j) {
      int e  = (tid + 256 * j) * 4;
      int nl = e & 127, kl = e >> 7;
      const float4 f = *(const float4*)(B + (size_t)(k0 + kl) * N + n0 + nl);
      Bt[(nl + 0) * 72 + kl] = (_Float16)f.x;
      Bt[(nl + 1) * 72 + kl] = (_Float16)f.y;
      Bt[(nl + 2) * 72 + kl] = (_Float16)f.z;
      Bt[(nl + 3) * 72 + kl] = (_Float16)f.w;
    }
    __syncthreads();

    // Two 16x16x32 WMMA stages per K-tile; A fragment reused for 2 B tiles.
#pragma unroll
    for (int kk = 0; kk < 64; kk += 32) {
      AFrag a, b0, b1;
      // A fragment: lanes 0-15 K=kk..kk+15,
      // lanes 16-31 K=kk+16..kk+31, row M = lane%16.
      const int mrow = wm * 16 + (lane & 15);
      const int kbA  = kk + ((lane & 16) ? 16 : 0);
      a.h[0] = *(const half8*)&As[mrow * 72 + kbA];
      a.h[1] = *(const half8*)&As[mrow * 72 + kbA + 8];
      // B fragments: col N = lane%16, lanes 0-15 K=kk..kk+15, lanes 16-31
      // K=kk+16..kk+31 (contiguous because Bt is N-major).
      const int nrow = wn * 32 + (lane & 15);
      const int kbB  = kk + ((lane & 16) ? 16 : 0);
      b0.h[0] = *(const half8*)&Bt[nrow * 72 + kbB];
      b0.h[1] = *(const half8*)&Bt[nrow * 72 + kbB + 8];
      b1.h[0] = *(const half8*)&Bt[(nrow + 16) * 72 + kbB];
      b1.h[1] = *(const half8*)&Bt[(nrow + 16) * 72 + kbB + 8];
      acc0 = __builtin_amdgcn_wmma_f32_16x16x32_f16(false, a.v, false, b0.v,
                                                    (short)0, acc0, false, false);
      acc1 = __builtin_amdgcn_wmma_f32_16x16x32_f16(false, a.v, false, b1.v,
                                                    (short)0, acc1, false, false);
    }
  }

  // Epilogue. C/D layout: VGPR r -> M = r + (lane<16?0:8), N = lane%16.
  const int nG = n0 + wn * 32 + (lane & 15);
  const float bv0 = bias ? bias[nG] : 0.0f;
  const float bv1 = bias ? bias[nG + 16] : 0.0f;
#pragma unroll
  for (int r = 0; r < 8; ++r) {
    const int mG = m0 + wm * 16 + r + ((lane & 16) ? 8 : 0);
    float v0 = (acc0[r] + bv0) * scale;
    float v1 = (acc1[r] + bv1) * scale;
    if (relu) { v0 = fmaxf(v0, 0.0f); v1 = fmaxf(v1, 0.0f); }
    if (resid) {
      v0 += resid[(size_t)mG * N + nG];
      v1 += resid[(size_t)mG * N + nG + 16];
    }
    C[(size_t)mG * N + nG]      = v0;
    C[(size_t)mG * N + nG + 16] = v1;
  }
}

// ---------------------------------------------------------------------------
// Elementwise helpers
// ---------------------------------------------------------------------------
__global__ void vadd(const float* __restrict__ a, const float* __restrict__ b,
                     float* __restrict__ c, size_t n)
{
  for (size_t i = (size_t)blockIdx.x * blockDim.x + threadIdx.x; i < n;
       i += (size_t)gridDim.x * blockDim.x)
    c[i] = a[i] + b[i];
}

__global__ void vcopy(const float* __restrict__ a, float* __restrict__ b, size_t n)
{
  for (size_t i = (size_t)blockIdx.x * blockDim.x + threadIdx.x; i < n;
       i += (size_t)gridDim.x * blockDim.x)
    b[i] = a[i];
}

__global__ __launch_bounds__(256)
void mask_rows(float* __restrict__ v, const int* __restrict__ pad)
{
  // zero row if padding flag set (val *= 1 - pad)
  if (pad[blockIdx.x] != 0) v[(size_t)blockIdx.x * E + threadIdx.x] = 0.0f;
}

// LayerNorm over E=256 (optionally x+resid first). One block per row.
__global__ __launch_bounds__(256)
void add_layernorm(const float* __restrict__ x, const float* __restrict__ resid,
                   const float* __restrict__ g, const float* __restrict__ bta,
                   float* __restrict__ y, int useResid)
{
  __shared__ float red[256];
  const size_t row = blockIdx.x;
  const int tid = threadIdx.x;
  float v = x[row * E + tid];
  if (useResid) v += resid[row * E + tid];
  red[tid] = v;
  __syncthreads();
  for (int st = 128; st > 0; st >>= 1) {
    if (tid < st) red[tid] += red[tid + st];
    __syncthreads();
  }
  const float mu = red[0] * (1.0f / 256.0f);
  __syncthreads();
  const float dv = v - mu;
  red[tid] = dv * dv;
  __syncthreads();
  for (int st = 128; st > 0; st >>= 1) {
    if (tid < st) red[tid] += red[tid + st];
    __syncthreads();
  }
  const float r = rsqrtf(red[0] * (1.0f / 256.0f) + 1e-5f);
  y[row * E + tid] = dv * r * g[tid] + bta[tid];
}

// ---------------------------------------------------------------------------
// Attention, many keys (v2l + lang self-attn). One block per (h,q) of batch b.
// Query row staged into LDS via global_load_async_to_lds_b128 (ASYNCcnt).
// ---------------------------------------------------------------------------
__global__ __launch_bounds__(256)
void attn_bigk(const float* __restrict__ qh,    // (bs,Sq,AD)
               const float* __restrict__ kh_b,  // (Sk,AD)
               const float* __restrict__ vh_b,  // (Sk,AD)
               const int* __restrict__ mask,    // (bs,Sk)
               int invertMask,
               float* __restrict__ o,           // (bs,Sq,AD)
               float* __restrict__ scoreWs,
               int b, int Sq, int Sk, int scStride)
{
  const int blk = blockIdx.x;
  const int h = blk / Sq, q = blk % Sq;
  const int tid = threadIdx.x;
  __shared__ float qrow[HD];
  __shared__ float red[256];

  // Async copy of the 512B query row (one wave, 16B per lane) into LDS.
  if (tid < 32) {
    const float* gp = qh + ((size_t)(b * Sq + q)) * AD + h * HD + tid * 4;
    unsigned ldsOff = (unsigned)(size_t)(&qrow[tid * 4]);
    unsigned long long ga = (unsigned long long)(size_t)gp;
    asm volatile("global_load_async_to_lds_b128 %0, %1, off"
                 :: "v"(ldsOff), "v"(ga) : "memory");
  }
  asm volatile("s_wait_asynccnt 0x0" ::: "memory");
  __syncthreads();

  float* srow = scoreWs + (size_t)blk * scStride;
  float lmax = -3.0e38f;
  for (int k = tid; k < Sk; k += 256) {
    const float* kp = kh_b + (size_t)k * AD + h * HD;
    float s = 0.0f;
    for (int d = 0; d < HD; ++d) s += qrow[d] * kp[d];
    s = fminf(fmaxf(s, -CLAMPV), CLAMPV);
    const int keep = ((mask[b * Sk + k] != 0) ? 1 : 0) ^ invertMask;
    if (!keep) s = -3.0e38f;
    srow[k] = s;
    lmax = fmaxf(lmax, s);
  }
  red[tid] = lmax;
  __syncthreads();
  for (int st = 128; st > 0; st >>= 1) {
    if (tid < st) red[tid] = fmaxf(red[tid], red[tid + st]);
    __syncthreads();
  }
  const float mx = red[0];
  __syncthreads();

  float lsum = 0.0f;
  for (int k = tid; k < Sk; k += 256) {
    const float s = srow[k];
    const float p = (s > -1.0e38f) ? __expf(s - mx) : 0.0f;
    srow[k] = p;
    lsum += p;
  }
  red[tid] = lsum;
  __syncthreads();
  for (int st = 128; st > 0; st >>= 1) {
    if (tid < st) red[tid] += red[tid + st];
    __syncthreads();
  }
  const float inv = 1.0f / red[0];

  if (tid < HD) {
    float acc = 0.0f;
    for (int k = 0; k < Sk; ++k)
      acc += srow[k] * vh_b[(size_t)k * AD + h * HD + tid];
    o[((size_t)(b * Sq + q)) * AD + h * HD + tid] = acc * inv;
  }
}

// ---------------------------------------------------------------------------
// Attention, few keys (l2v: 5440 queries x 20 keys). One block per query.
// ---------------------------------------------------------------------------
__global__ __launch_bounds__(256)
void attn_smallk(const float* __restrict__ qh_b, // (Sq,AD), batch slice
                 const float* __restrict__ kh,   // (bs,Sk,AD)
                 const float* __restrict__ vh,   // (bs,Sk,AD)
                 const int* __restrict__ mask,   // (bs,Sk)
                 float* __restrict__ o_b,        // (Sq,AD), batch slice
                 int b, int Sk)
{
  const int q = blockIdx.x;
  const int tid = threadIdx.x;
  __shared__ float qrow[HD];
  __shared__ float sc[32];
  for (int h = 0; h < NH; ++h) {
    __syncthreads();
    if (tid < HD) qrow[tid] = qh_b[(size_t)q * AD + h * HD + tid];
    __syncthreads();
    if (tid < Sk) {
      const float* kp = kh + ((size_t)(b * Sk + tid)) * AD + h * HD;
      float s = 0.0f;
      for (int d = 0; d < HD; ++d) s += qrow[d] * kp[d];
      s = fminf(fmaxf(s, -CLAMPV), CLAMPV);
      if (mask[b * Sk + tid] == 0) s = -3.0e38f;
      sc[tid] = s;
    }
    __syncthreads();
    if (tid == 0) {
      float mx = -3.0e38f;
      for (int k = 0; k < Sk; ++k) mx = fmaxf(mx, sc[k]);
      float sm = 0.0f;
      for (int k = 0; k < Sk; ++k) {
        const float p = (sc[k] > -1.0e38f) ? __expf(sc[k] - mx) : 0.0f;
        sc[k] = p; sm += p;
      }
      const float inv = 1.0f / sm;
      for (int k = 0; k < Sk; ++k) sc[k] *= inv;
    }
    __syncthreads();
    if (tid < HD) {
      float acc = 0.0f;
      for (int k = 0; k < Sk; ++k)
        acc += sc[k] * vh[((size_t)(b * Sk + k)) * AD + h * HD + tid];
      o_b[(size_t)q * AD + h * HD + tid] = acc;
    }
  }
}

// ---------------------------------------------------------------------------
// Deformable attention: per-(h) 16-way softmax, then bilinear sampling.
// ---------------------------------------------------------------------------
__global__ __launch_bounds__(128)
void aw_softmax(float* __restrict__ aw)   // (bs*L, 128), softmax over 16 per head
{
  __shared__ float s[128];
  __shared__ float mxs[8], sms[8];
  const size_t row = blockIdx.x;
  const int t = threadIdx.x;
  const int h = t >> 4, j = t & 15;
  s[t] = aw[row * 128 + t];
  __syncthreads();
  if (j == 0) {
    float mx = -3.0e38f;
    for (int i = 0; i < 16; ++i) mx = fmaxf(mx, s[h * 16 + i]);
    float sm = 0.0f;
    for (int i = 0; i < 16; ++i) sm += __expf(s[h * 16 + i] - mx);
    mxs[h] = mx; sms[h] = sm;
  }
  __syncthreads();
  aw[row * 128 + t] = __expf(s[t] - mxs[h]) / sms[h];
}

__global__ __launch_bounds__(256)
void deform_sample(const float* __restrict__ val,  // (bs*L, E), masked
                   const float* __restrict__ off,  // (bs*L, E) = (h,lvl,p,2)
                   const float* __restrict__ aw,   // (bs*L, 128)
                   const float* __restrict__ ref,  // (bs*L, 4, 2)
                   float* __restrict__ out)        // (bs*L, E)
{
  const size_t rowg = blockIdx.x;          // b*L + l
  const int b = (int)(rowg / LTOT);
  const int tid = threadIdx.x;
  const int h = tid >> 5, d = tid & 31;
  const int starts[4] = {0, 4096, 5120, 5376};
  const int DIMS[4]   = {64, 32, 16, 8};   // square levels: Hl == Wl
  float acc = 0.0f;
#pragma unroll
  for (int lvl = 0; lvl < 4; ++lvl) {
    const float rx = ref[rowg * 8 + lvl * 2 + 0];
    const float ry = ref[rowg * 8 + lvl * 2 + 1];
    const int Hl = DIMS[lvl], Wl = DIMS[lvl], st = starts[lvl];
    const float fW = (float)Wl, fH = (float)Hl;
#pragma unroll
    for (int p = 0; p < 4; ++p) {
      const int oidx = ((h * 4 + lvl) * 4 + p) * 2;
      const float ox = off[rowg * E + oidx + 0];
      const float oy = off[rowg * E + oidx + 1];
      const float w  = aw[rowg * 128 + (h * 4 + lvl) * 4 + p];
      const float x = (rx + ox / fW) * fW - 0.5f;
      const float y = (ry + oy / fH) * fH - 0.5f;
      const float x0f = floorf(x), y0f = floorf(y);
      const int x0 = (int)x0f, y0 = (int)y0f;
      const float wx = x - x0f, wy = y - y0f;
      float s = 0.0f;
#pragma unroll
      for (int dy = 0; dy < 2; ++dy) {
#pragma unroll
        for (int dx = 0; dx < 2; ++dx) {
          const int ix = x0 + dx, iy = y0 + dy;
          if (ix >= 0 && ix < Wl && iy >= 0 && iy < Hl) {
            const float wgt = (dx ? wx : 1.0f - wx) * (dy ? wy : 1.0f - wy);
            const size_t pos = (size_t)b * LTOT + st + iy * Wl + ix;
            s += wgt * val[pos * E + h * VHD + d];
          }
        }
      }
      acc += w * s;
    }
  }
  out[rowg * E + tid] = acc;
}

// ---------------------------------------------------------------------------
// Host-side orchestration
// ---------------------------------------------------------------------------
static inline void gemm(hipStream_t s, const float* A, const float* B,
                        const float* bias, const float* resid, float* C,
                        int M, int N, int K, float scale = 1.0f, int relu = 0)
{
  dim3 g(M / 32, N / 128);
  gemm_f16_wmma<<<g, 256, 0, s>>>(A, B, bias, resid, C, M, N, K, scale, relu);
}

extern "C" void kernel_launch(void* const* d_in, const int* in_sizes, int n_in,
                              void* d_out, int out_size, void* d_ws, size_t ws_size,
                              hipStream_t stream)
{
  (void)in_sizes; (void)n_in; (void)out_size; (void)ws_size;
  const float* vis       = (const float*)d_in[0];
  const float* lang      = (const float*)d_in[1];
  const int*   lang_mask = (const int*)d_in[2];
  const float* vis_pos   = (const float*)d_in[3];
  const int*   pad       = (const int*)d_in[4];
  const float* refp      = (const float*)d_in[5];
  // d_in[6] spatial_shapes, d_in[7] level_start_index: compile-time constants.

  const float** P = (const float**)&d_in[8];
  const float *v2l_wq = P[0],  *v2l_bq = P[1],  *v2l_wk = P[2],  *v2l_bk = P[3];
  const float *v2l_wv = P[4],  *v2l_bv = P[5],  *v2l_wo = P[6],  *v2l_bo = P[7];
  const float *l2v_wq = P[8],  *l2v_bq = P[9],  *l2v_wk = P[10], *l2v_bk = P[11];
  const float *l2v_wv = P[12], *l2v_bv = P[13], *l2v_wo = P[14], *l2v_bo = P[15];
  const float *ls_wq  = P[16], *ls_bq  = P[17], *ls_wk  = P[18], *ls_bk  = P[19];
  const float *ls_wv  = P[20], *ls_bv  = P[21], *ls_wo  = P[22], *ls_bo  = P[23];
  const float *d_wval = P[24], *d_bval = P[25], *d_woff = P[26], *d_boff = P[27];
  const float *d_ww   = P[28], *d_bw   = P[29], *d_wout = P[30], *d_bout = P[31];
  const float *v_w1   = P[32], *v_b1   = P[33], *v_w2   = P[34], *v_b2   = P[35];
  const float *l_w1   = P[36], *l_b1   = P[37], *l_w2   = P[38], *l_b2   = P[39];
  const float *vn1g = P[40], *vn1b = P[41], *vn2g = P[42], *vn2b = P[43];
  const float *vn3g = P[44], *vn3b = P[45], *ln1g = P[46], *ln1b = P[47];
  const float *ln2g = P[48], *ln2b = P[49], *ln3g = P[50], *ln3b = P[51];

  const size_t MV = (size_t)BS * LTOT;        // 43520 vision rows
  const size_t ML = (size_t)BS * SQL;         // 160 language rows
  const size_t SLICE_E  = (size_t)LTOT * E;   // per-batch vision E block
  const size_t SLICE_AD = (size_t)LTOT * AD;

  // Workspace carving (~250 MB of fp32).
  float* W = (float*)d_ws;
  size_t off = 0;
  auto take = [&](size_t n) { float* p = W + off; off += n; return p; };
  float* wA   = take(SLICE_AD);       // per-batch AD scratch (kh / q / hidden)
  float* wB   = take(SLICE_AD);       // per-batch AD scratch (vh / attn out)
  float* wC   = take(MV * E);         // vis+vis_pos  -> deform val
  float* wD   = take(MV * E);         // vis2 / vis3
  float* wE   = take(MV * E);         // l2v out / qd / sampled
  float* wF   = take(MV * E);         // off / deform out / ffn out
  float* wSC  = take(MV * 128);       // score rows / attention weights
  float* qh_l = take(ML * AD);
  float* o_l  = take(ML * AD);
  float* kh_l = take(ML * AD);
  float* vh_l = take(ML * AD);
  float* plA  = take(ML * E);
  float* tl   = take(ML * E);
  float* tl2  = take(ML * E);

  // ---- Step 0: buf_vp = vis + vis_pos -----------------------------------
  vadd<<<4096, 256, 0, stream>>>(vis, vis_pos, wC, MV * E);

  // ---- v2l cross attention (q=lang, k=vis+pos, v=vis) -------------------
  gemm(stream, lang, v2l_wq, v2l_bq, nullptr, qh_l, (int)ML, AD, E, QSCALE);
  for (int b = 0; b < BS; ++b) {
    gemm(stream, wC + b * SLICE_E, v2l_wk, v2l_bk, nullptr, wA, LTOT, AD, E);
    gemm(stream, vis + b * SLICE_E, v2l_wv, v2l_bv, nullptr, wB, LTOT, AD, E);
    attn_bigk<<<NH * SQL, 256, 0, stream>>>(qh_l, wA, wB, pad, /*invert*/1,
                                            o_l, wSC, b, SQL, LTOT, LTOT);
  }
  gemm(stream, o_l, v2l_wo, v2l_bo, nullptr, tl, (int)ML, E, AD);
  add_layernorm<<<(int)ML, 256, 0, stream>>>(tl, lang, ln1g, ln1b, plA, 1);

  // ---- language self attention ------------------------------------------
  gemm(stream, plA, ls_wq, ls_bq, nullptr, qh_l, (int)ML, AD, E, QSCALE);
  gemm(stream, plA, ls_wk, ls_bk, nullptr, kh_l, (int)ML, AD, E);
  gemm(stream, plA, ls_wv, ls_bv, nullptr, vh_l, (int)ML, AD, E);
  for (int b = 0; b < BS; ++b)
    attn_bigk<<<NH * SQL, 256, 0, stream>>>(qh_l, kh_l + (size_t)b * SQL * AD,
                                            vh_l + (size_t)b * SQL * AD,
                                            lang_mask, 0, o_l, wSC, b, SQL, SQL, 32);
  gemm(stream, o_l, ls_wo, ls_bo, nullptr, tl, (int)ML, E, AD);
  add_layernorm<<<(int)ML, 256, 0, stream>>>(tl, plA, ln2g, ln2b, tl2, 1);

  // ---- language FFN ------------------------------------------------------
  gemm(stream, tl2, l_w1, l_b1, nullptr, qh_l, (int)ML, AD, E, 1.0f, /*relu*/1);
  gemm(stream, qh_l, l_w2, l_b2, nullptr, tl, (int)ML, E, AD);
  add_layernorm<<<(int)ML, 256, 0, stream>>>(tl, tl2, ln3g, ln3b, plA, 1);  // pl final

  // ---- l2v cross attention (q=vis+pos, k=v=pl) --------------------------
  gemm(stream, plA, l2v_wk, l2v_bk, nullptr, kh_l, (int)ML, AD, E);
  gemm(stream, plA, l2v_wv, l2v_bv, nullptr, vh_l, (int)ML, AD, E);
  for (int b = 0; b < BS; ++b) {
    gemm(stream, wC + b * SLICE_E, l2v_wq, l2v_bq, nullptr, wA, LTOT, AD, E, QSCALE);
    attn_smallk<<<LTOT, 256, 0, stream>>>(wA, kh_l, vh_l, lang_mask, wB, b, SQL);
    gemm(stream, wB, l2v_wo, l2v_bo, nullptr, wE + b * SLICE_E, LTOT, E, AD);
  }
  add_layernorm<<<(int)MV, 256, 0, stream>>>(wE, vis, vn2g, vn2b, wD, 1);   // vis2

  // ---- multi-scale deformable attention ---------------------------------
  vadd<<<4096, 256, 0, stream>>>(wD, vis_pos, wE, MV * E);                  // qd
  gemm(stream, wD, d_wval, d_bval, nullptr, wC, (int)MV, E, E);             // val
  mask_rows<<<(int)MV, 256, 0, stream>>>(wC, pad);
  gemm(stream, wE, d_woff, d_boff, nullptr, wF, (int)MV, E, E);             // offsets
  gemm(stream, wE, d_ww, d_bw, nullptr, wSC, (int)MV, 128, E);              // aw logits
  aw_softmax<<<(int)MV, 128, 0, stream>>>(wSC);
  deform_sample<<<(int)MV, 256, 0, stream>>>(wC, wF, wSC, refp, wE);        // sampled
  gemm(stream, wE, d_wout, d_bout, wD /*identity*/, wF, (int)MV, E, E);     // +resid
  add_layernorm<<<(int)MV, 256, 0, stream>>>(wF, nullptr, vn1g, vn1b, wD, 0); // vis3

  // ---- vision FFN ---------------------------------------------------------
  for (int b = 0; b < BS; ++b) {
    gemm(stream, wD + b * SLICE_E, v_w1, v_b1, nullptr, wA, LTOT, AD, E, 1.0f, 1);
    gemm(stream, wA, v_w2, v_b2, nullptr, wF + b * SLICE_E, LTOT, E, AD);
  }
  float* out_vis = (float*)d_out;
  add_layernorm<<<(int)MV, 256, 0, stream>>>(wF, wD, vn3g, vn3b, out_vis, 1);

  // ---- append final language stream to output ----------------------------
  vcopy<<<64, 256, 0, stream>>>(plA, out_vis + MV * E, ML * E);
}
